// GroupedQueryAttention_19988777796228
// MI455X (gfx1250) — matrix operand order repaired and checked by reading, not compiled, not run on hardware
//
#include <hip/hip_runtime.h>

// ---------------- types ----------------
typedef __bf16 bf16;
typedef __bf16  v16bf  __attribute__((ext_vector_type(16)));
typedef __bf16  bf16x8 __attribute__((ext_vector_type(8)));
typedef __bf16  bf16x4 __attribute__((ext_vector_type(4)));
typedef float   v8f    __attribute__((ext_vector_type(8)));

union BF16Frag { v16bf v; bf16x8 h[2]; };

// ---------------- async global->LDS (gfx1250), with sync fallback ----------------
#if __has_builtin(__builtin_amdgcn_global_load_async_to_lds_b128)
#define USE_ASYNC_LDS 1
#define GLOBAL_AS __attribute__((address_space(1)))
#define LDS_AS    __attribute__((address_space(3)))
typedef int async_b128_t __attribute__((vector_size(4 * sizeof(int))));
__device__ __forceinline__ void async_b128(const bf16* g, bf16* l) {
    __builtin_amdgcn_global_load_async_to_lds_b128(
        (GLOBAL_AS async_b128_t*)(bf16*)g, (LDS_AS async_b128_t*)l, 0, 0);
}
#if __has_builtin(__builtin_amdgcn_s_wait_asynccnt)
#define WAIT_ASYNC(n) __builtin_amdgcn_s_wait_asynccnt(n)
#else
#define WAIT_ASYNC(n) asm volatile("s_wait_asynccnt %0" ::"i"(n) : "memory")
#endif
#endif

#define SWZ(x, imm) __int_as_float(__builtin_amdgcn_ds_swizzle(__float_as_int(x), imm))

__device__ inline float redmax16(float x) {
    x = fmaxf(x, SWZ(x, 0x041F));   // xor 1
    x = fmaxf(x, SWZ(x, 0x081F));   // xor 2
    x = fmaxf(x, SWZ(x, 0x101F));   // xor 4
    x = fmaxf(x, SWZ(x, 0x201F));   // xor 8
    return x;
}
__device__ inline float redsum16(float x) {
    x += SWZ(x, 0x041F);
    x += SWZ(x, 0x081F);
    x += SWZ(x, 0x101F);
    x += SWZ(x, 0x201F);
    return x;
}

// ---------------- constants ----------------
#define S_LEN   2048
#define DMODEL  4096
#define HQ      32
#define HKV     8
#define HD      128
#define KV_DIM  1024

// ---------------- f32 -> bf16 cast (x4 vectorized) ----------------
__global__ __launch_bounds__(256) void cast_f32_bf16_x4(const float4* __restrict__ src,
                                                        bf16* __restrict__ dst, int n4) {
    int i = blockIdx.x * blockDim.x + threadIdx.x;
    if (i < n4) {
        float4 v = src[i];
        bf16x4 o;
        o[0] = (bf16)v.x; o[1] = (bf16)v.y; o[2] = (bf16)v.z; o[3] = (bf16)v.w;
        *(bf16x4*)&dst[4 * i] = o;
    }
}

// ---------------- bf16 GEMM: C[M,N] = A[M,K] * B[N,K]^T ----------------
// 256 threads = 8 waves; tile 128x128, K-step 32; wave = 32x64 (2x4 wmma tiles)
// Double-buffered LDS; async global->LDS staging overlaps with WMMA.
#define GT_M 128
#define GT_N 128
#define GT_K 32
#define LDT  40   // padded LDS row stride (bf16 elems); 80B = 16B-aligned rows

__global__ __launch_bounds__(256)
void gemm_bf16_nt(const bf16* __restrict__ A, const bf16* __restrict__ B,
                  float* __restrict__ Cf, bf16* __restrict__ Cb,
                  int M, int N, int K) {
    __shared__ bf16 As[2][GT_M * LDT];
    __shared__ bf16 Bs[2][GT_N * LDT];

    const int nTilesN = N / GT_N;
    const int bm = (int)blockIdx.x / nTilesN;
    const int bn = (int)blockIdx.x % nTilesN;
    const int tid  = threadIdx.x;
    const int lane = tid & 31;
    const int wave = tid >> 5;
    const int wm = (wave & 3) * 32;
    const int wn = (wave >> 2) * 64;
    const int half = lane >> 4;
    const int l16  = lane & 15;

    v8f zero = {0.f, 0.f, 0.f, 0.f, 0.f, 0.f, 0.f, 0.f};
    v8f acc[2][4];
#pragma unroll
    for (int i = 0; i < 2; ++i)
#pragma unroll
        for (int j = 0; j < 4; ++j) acc[i][j] = zero;

    // stage one K-tile (4 x b128 per thread: 2 for A, 2 for B)
    auto stage = [&](int sbuf, int k0) {
#pragma unroll
        for (int i = 0; i < 2; ++i) {
            int c  = tid + i * 256;
            int r  = c >> 2;
            int cc = (c & 3) * 8;
            const bf16* ga = &A[(size_t)(bm * GT_M + r) * K + k0 + cc];
            const bf16* gb = &B[(size_t)(bn * GT_N + r) * K + k0 + cc];
#ifdef USE_ASYNC_LDS
            async_b128(ga, &As[sbuf][r * LDT + cc]);
            async_b128(gb, &Bs[sbuf][r * LDT + cc]);
#else
            *(bf16x8*)&As[sbuf][r * LDT + cc] = *(const bf16x8*)ga;
            *(bf16x8*)&Bs[sbuf][r * LDT + cc] = *(const bf16x8*)gb;
#endif
        }
    };

    stage(0, 0);
    const int kLast = K - GT_K;
    int buf = 0;
    for (int k0 = 0; k0 < K; k0 += GT_K, buf ^= 1) {
        __syncthreads();                       // everyone done reading buf^1
        // prefetch next tile (clamped: last iter re-reads last tile, never consumed)
        int kn = k0 + GT_K;
        stage(buf ^ 1, kn > kLast ? kLast : kn);
#ifdef USE_ASYNC_LDS
        WAIT_ASYNC(4);                         // <=4 outstanding == next tile only
#endif
        __syncthreads();                       // all waves' pieces of buf present

        BF16Frag af[2], bfr[4];
#pragma unroll
        for (int i = 0; i < 2; ++i) {
            const bf16* p = &As[buf][(wm + i * 16 + l16) * LDT];
            af[i].h[0] = *(const bf16x8*)(p + 8 * half);        // K 0..7 | 8..15
            af[i].h[1] = *(const bf16x8*)(p + 16 + 8 * half);   // K 16..23 | 24..31
        }
#pragma unroll
        for (int j = 0; j < 4; ++j) {
            const bf16* p = &Bs[buf][(wn + j * 16 + l16) * LDT + 8 * half];
            bfr[j].h[0] = *(const bf16x8*)(p);
            bfr[j].h[1] = *(const bf16x8*)(p + 16);
        }
#pragma unroll
        for (int i = 0; i < 2; ++i)
#pragma unroll
            for (int j = 0; j < 4; ++j)
                acc[i][j] = __builtin_amdgcn_wmma_f32_16x16x32_bf16(
                    false, af[i].v, false, bfr[j].v, (short)0, acc[i][j], false, false);
    }

#pragma unroll
    for (int i = 0; i < 2; ++i)
#pragma unroll
        for (int j = 0; j < 4; ++j)
#pragma unroll
            for (int r = 0; r < 8; ++r) {
                int row = bm * GT_M + wm + i * 16 + half * 8 + r;
                int col = bn * GT_N + wn + j * 16 + l16;
                float v = acc[i][j][r];
                if (Cf) Cf[(size_t)row * N + col] = v;
                if (Cb) Cb[(size_t)row * N + col] = (bf16)v;
            }
}

// ---------------- flash attention ----------------
// grid: (S/64) * HQ blocks, 128 threads = 4 waves; wave owns 16 q rows.
#define AQ  64
#define AKB 64
#define QLD 136   // 128 + 8 pad (272B rows, 16B-aligned)
#define VLD 72    // 64 + 8 pad

__global__ __launch_bounds__(128)
void attn_flash(const bf16* __restrict__ Q, const bf16* __restrict__ K,
                const bf16* __restrict__ V, bf16* __restrict__ O, float scale) {
    __shared__ bf16 Qs[AQ * QLD];        // 64 x 128
    __shared__ bf16 Ks[AKB * QLD];       // 64 x 128
    __shared__ bf16 Vt[HD * VLD];        // transposed: [d][key]
    __shared__ bf16 Ps[4][16 * AKB];     // per-wave P tile

    const int h   = (int)blockIdx.x % HQ;
    const int qb  = (int)blockIdx.x / HQ;
    const int kvh = h >> 2;              // repeat_interleave mapping
    const int tid  = threadIdx.x;
    const int lane = tid & 31;
    const int wave = tid >> 5;
    const int half = lane >> 4;
    const int l16  = lane & 15;

    // load Q block (64 x 128): 1024 chunks / 128 threads (async if available)
#pragma unroll
    for (int i = 0; i < 8; ++i) {
        int c = tid + i * 128;
        int r = c >> 4, cc = (c & 15) * 8;
        const bf16* gq = &Q[(size_t)(qb * AQ + r) * DMODEL + h * HD + cc];
#ifdef USE_ASYNC_LDS
        async_b128(gq, &Qs[r * QLD + cc]);
#else
        *(bf16x8*)&Qs[r * QLD + cc] = *(const bf16x8*)gq;
#endif
    }

    v8f zero = {0.f, 0.f, 0.f, 0.f, 0.f, 0.f, 0.f, 0.f};
    v8f o[8];
    float rm[8], rl[8];
#pragma unroll
    for (int j = 0; j < 8; ++j) { o[j] = zero; rm[j] = -1e30f; rl[j] = 0.f; }

    for (int kb = 0; kb < S_LEN / AKB; ++kb) {
        __syncthreads();
        // stage K tile (row-major, async) and V tile (transposed, via VGPRs)
#pragma unroll
        for (int i = 0; i < 8; ++i) {
            int c = tid + i * 128;
            int r = c >> 4, cc = (c & 15) * 8;
            const bf16* gk = &K[(size_t)(kb * AKB + r) * KV_DIM + kvh * HD + cc];
#ifdef USE_ASYNC_LDS
            async_b128(gk, &Ks[r * QLD + cc]);
#else
            *(bf16x8*)&Ks[r * QLD + cc] = *(const bf16x8*)gk;
#endif
            bf16x8 vv = *(const bf16x8*)&V[(size_t)(kb * AKB + r) * KV_DIM + kvh * HD + cc];
#pragma unroll
            for (int t = 0; t < 8; ++t) Vt[(cc + t) * VLD + r] = vv[t];
        }
#ifdef USE_ASYNC_LDS
        WAIT_ASYNC(0);   // K tile (and Q on first iter) landed
#endif
        __syncthreads();

        // scores S[16 q, 64 keys] for this wave
        v8f s[4];
#pragma unroll
        for (int j = 0; j < 4; ++j) s[j] = zero;
#pragma unroll
        for (int d0 = 0; d0 < HD; d0 += 32) {
            BF16Frag aq;
            const bf16* p = &Qs[(wave * 16 + l16) * QLD + d0];
            aq.h[0] = *(const bf16x8*)(p + 8 * half);
            aq.h[1] = *(const bf16x8*)(p + 16 + 8 * half);
#pragma unroll
            for (int j = 0; j < 4; ++j) {
                BF16Frag bk;
                const bf16* q = &Ks[(j * 16 + l16) * QLD + d0 + 8 * half];
                bk.h[0] = *(const bf16x8*)(q);
                bk.h[1] = *(const bf16x8*)(q + 16);
                s[j] = __builtin_amdgcn_wmma_f32_16x16x32_bf16(
                    false, aq.v, false, bk.v, (short)0, s[j], false, false);
            }
        }

        // online softmax: row stats live per (vgpr r, 16-lane half)
        float pm[8], corr[8], psum[8];
#pragma unroll
        for (int r = 0; r < 8; ++r) {
            float t = s[0][r];
            t = fmaxf(t, s[1][r]); t = fmaxf(t, s[2][r]); t = fmaxf(t, s[3][r]);
            pm[r] = redmax16(t * scale);
            float nm = fmaxf(rm[r], pm[r]);
            corr[r] = __expf(rm[r] - nm);
            rm[r] = nm;
            rl[r] *= corr[r];
            psum[r] = 0.f;
        }
#pragma unroll
        for (int j = 0; j < 4; ++j)
#pragma unroll
            for (int r = 0; r < 8; ++r) {
                float p = __expf(s[j][r] * scale - rm[r]);
                psum[r] += p;
                Ps[wave][(half * 8 + r) * AKB + j * 16 + l16] = (bf16)p;
            }
#pragma unroll
        for (int r = 0; r < 8; ++r) rl[r] += redsum16(psum[r]);
#pragma unroll
        for (int jt = 0; jt < 8; ++jt)
#pragma unroll
            for (int r = 0; r < 8; ++r) o[jt][r] *= corr[r];

        __builtin_amdgcn_fence(__ATOMIC_SEQ_CST, "wavefront");  // P visible across lanes

        // O += P * V   (A = P[16 x 64 keys], B = Vt columns)
#pragma unroll
        for (int kc = 0; kc < 2; ++kc) {
            BF16Frag ap;
            const bf16* p = &Ps[wave][l16 * AKB + kc * 32];
            ap.h[0] = *(const bf16x8*)(p + 8 * half);
            ap.h[1] = *(const bf16x8*)(p + 16 + 8 * half);
#pragma unroll
            for (int jt = 0; jt < 8; ++jt) {
                BF16Frag bv;
                const bf16* q = &Vt[(jt * 16 + l16) * VLD + kc * 32 + 8 * half];
                bv.h[0] = *(const bf16x8*)(q);
                bv.h[1] = *(const bf16x8*)(q + 16);
                o[jt] = __builtin_amdgcn_wmma_f32_16x16x32_bf16(
                    false, ap.v, false, bv.v, (short)0, o[jt], false, false);
            }
        }
    }

    // normalize and store bf16 attention output [S, DMODEL] head-major columns
#pragma unroll
    for (int r = 0; r < 8; ++r) rl[r] = 1.0f / rl[r];
#pragma unroll
    for (int jt = 0; jt < 8; ++jt)
#pragma unroll
        for (int r = 0; r < 8; ++r) {
            int row = qb * AQ + wave * 16 + half * 8 + r;
            int col = h * HD + jt * 16 + l16;
            O[(size_t)row * DMODEL + col] = (bf16)(o[jt][r] * rl[r]);
        }
}

// ---------------- kv cache copy (repeat_interleave heads [0,0,0,0,1,1,1,1]) ----------------
__global__ __launch_bounds__(256)
void cache_copy(const bf16* __restrict__ Kb, const bf16* __restrict__ Vb,
                float* __restrict__ kout, float* __restrict__ vout, int n) {
    int i = blockIdx.x * blockDim.x + threadIdx.x;
    if (i >= n) return;
    int d = i & (HD - 1);
    int rest = i >> 7;
    int srow = rest & (S_LEN - 1);
    int ch = rest >> 11;            // cache head 0..7
    int kv = ch >> 2;               // -> kv head 0 or 1
    size_t src = (size_t)srow * KV_DIM + kv * HD + d;
    kout[i] = (float)Kb[src];
    vout[i] = (float)Vb[src];
}

// ---------------- launch ----------------
extern "C" void kernel_launch(void* const* d_in, const int* in_sizes, int n_in,
                              void* d_out, int out_size, void* d_ws, size_t ws_size,
                              hipStream_t stream) {
    const float* x  = (const float*)d_in[0];
    const float* Wq = (const float*)d_in[1];
    const float* Wk = (const float*)d_in[2];
    const float* Wv = (const float*)d_in[3];
    const float* Wo = (const float*)d_in[4];

    bf16* w = (bf16*)d_ws;
    bf16* xb    = w;                                // 8,388,608
    bf16* Wqb   = xb   + (size_t)S_LEN * DMODEL;    // 16,777,216
    bf16* Wkb   = Wqb  + (size_t)DMODEL * DMODEL;   //  4,194,304
    bf16* Wvb   = Wkb  + (size_t)KV_DIM * DMODEL;   //  4,194,304
    bf16* Wob   = Wvb  + (size_t)KV_DIM * DMODEL;   // 16,777,216
    bf16* Qb    = Wob  + (size_t)DMODEL * DMODEL;   //  8,388,608
    bf16* Kb    = Qb   + (size_t)S_LEN * DMODEL;    //  2,097,152
    bf16* Vb    = Kb   + (size_t)S_LEN * KV_DIM;    //  2,097,152
    bf16* Attnb = Vb   + (size_t)S_LEN * KV_DIM;    //  8,388,608

    float* out    = (float*)d_out;
    float* kcache = out + (size_t)S_LEN * DMODEL;
    float* vcache = kcache + (size_t)HKV * S_LEN * HD;

    auto cast = [&](const float* s, bf16* d, int n) {
        int n4 = n / 4;
        cast_f32_bf16_x4<<<(n4 + 255) / 256, 256, 0, stream>>>((const float4*)s, d, n4);
    };
    cast(x,  xb,  S_LEN * DMODEL);
    cast(Wq, Wqb, DMODEL * DMODEL);
    cast(Wk, Wkb, KV_DIM * DMODEL);
    cast(Wv, Wvb, KV_DIM * DMODEL);
    cast(Wo, Wob, DMODEL * DMODEL);

    // Q = x @ Wq^T ; K = x @ Wk^T ; V = x @ Wv^T
    gemm_bf16_nt<<<(S_LEN / GT_M) * (DMODEL / GT_N), 256, 0, stream>>>(
        xb, Wqb, nullptr, Qb, S_LEN, DMODEL, DMODEL);
    gemm_bf16_nt<<<(S_LEN / GT_M) * (KV_DIM / GT_N), 256, 0, stream>>>(
        xb, Wkb, nullptr, Kb, S_LEN, KV_DIM, DMODEL);
    gemm_bf16_nt<<<(S_LEN / GT_M) * (KV_DIM / GT_N), 256, 0, stream>>>(
        xb, Wvb, nullptr, Vb, S_LEN, KV_DIM, DMODEL);

    // attention
    float scale = 0.08838834764831845f;  // 1/sqrt(128)
    attn_flash<<<(S_LEN / AQ) * HQ, 128, 0, stream>>>(Qb, Kb, Vb, Attnb, scale);

    // out = attn @ Wo^T  (f32 output)
    gemm_bf16_nt<<<(S_LEN / GT_M) * (DMODEL / GT_N), 256, 0, stream>>>(
        Attnb, Wob, out, nullptr, S_LEN, DMODEL, DMODEL);

    // kv cache outputs
    int ncache = HKV * S_LEN * HD;
    cache_copy<<<(ncache + 255) / 256, 256, 0, stream>>>(Kb, Vb, kcache, vcache, ncache);
}
